// DeformableAttention_18717467476521
// MI455X (gfx1250) — matrix operand order repaired and checked by reading, not compiled, not run on hardware
//
#include <hip/hip_runtime.h>

// ---------------------------------------------------------------------------
// Global attention (Swin-style, bias path inactive) for MI455X / gfx1250.
//   1) cvt: x, w_qkv, w_proj  fp32 -> f16
//   2) gemm_wmma<true> : qkv_f16[16384,2304] = x_f16 @ w_qkv^T + b_qkv
//   3) attn_wmma       : flash-style online-softmax attention, f16 out
//   4) gemm_wmma<false>: d_out fp32 = attn_f16 @ w_proj^T + b_proj
// Matmuls: v_wmma_f32_16x16x32_f16 (wave32, fp32 accumulate).
// Tile staging: global_load_async_to_lds_b128 (ASYNCcnt) when available,
// with double-buffered LDS in the GEMMs to overlap copy and WMMA.
// ---------------------------------------------------------------------------

typedef _Float16 h16;
typedef __attribute__((ext_vector_type(16))) _Float16 v16h;
typedef __attribute__((ext_vector_type(8)))  _Float16 v8h;
typedef __attribute__((ext_vector_type(8)))  float    v8f;
// Matches the builtin's parameter type exactly (per hipcc diagnostic):
//   '__attribute__((__vector_size__(4 * sizeof(int)))) int __device__ *'
typedef __attribute__((vector_size(4 * sizeof(int)))) int v4i_b128;

#define WMMA_F32_F16(a, b, c)                                                  \
  __builtin_amdgcn_wmma_f32_16x16x32_f16(false, (a), false, (b), (short)0,     \
                                         (c), false, false)

// ---- CDNA5 async global->LDS copy (ASYNCcnt), guarded ----------------------
#if defined(__has_builtin)
#if __has_builtin(__builtin_amdgcn_global_load_async_to_lds_b128) && \
    __has_builtin(__builtin_amdgcn_s_wait_asynccnt)
#define USE_ASYNC_LDS 1
#endif
#endif
#ifndef USE_ASYNC_LDS
#define USE_ASYNC_LDS 0
#endif

#define AS_GLOBAL __attribute__((address_space(1)))
#define AS_LOCAL  __attribute__((address_space(3)))

static __device__ __forceinline__ void copy_b128_to_lds(const h16* g, h16* l) {
#if USE_ASYNC_LDS
  __builtin_amdgcn_global_load_async_to_lds_b128(
      (AS_GLOBAL v4i_b128*)g, (AS_LOCAL v4i_b128*)l, 0, 0);
#else
  *(v8h*)l = *(const v8h*)g;
#endif
}
static __device__ __forceinline__ void wait_async_copies() {
#if USE_ASYNC_LDS
  __builtin_amdgcn_s_wait_asynccnt(0);
#endif
}

// A-fragment (16x32, f16), ISA 7.12.2 layout:
//   lane L: row = L&15; elems 0..7 = K[koff..koff+7], elems 8..15 = K[16+koff..]
//   koff = (L<16) ? 0 : 8
static __device__ __forceinline__ v16h load_frag_a(const h16* base, int stride) {
  const int lane = threadIdx.x & 31;
  const h16* p = base + (lane & 15) * stride + ((lane & 16) ? 8 : 0);
  v8h lo = *(const v8h*)p;
  v8h hi = *(const v8h*)(p + 16);
  return __builtin_shufflevector(lo, hi, 0, 1, 2, 3, 4, 5, 6, 7, 8, 9, 10, 11,
                                 12, 13, 14, 15);
}

// B-fragment (32x16, f16) from B^T rows (i.e. [n][k] storage):
//   lane L: col n = L&15; elems 0..7 = K[koff..koff+7], elems 8..15 = K[16+koff..], koff = (L<16)?0:8
static __device__ __forceinline__ v16h load_frag_b(const h16* base, int stride) {
  const int lane = threadIdx.x & 31;
  const h16* p = base + (lane & 15) * stride + ((lane & 16) ? 8 : 0);
  v8h lo = *(const v8h*)p;
  v8h hi = *(const v8h*)(p + 16);
  return __builtin_shufflevector(lo, hi, 0, 1, 2, 3, 4, 5, 6, 7, 8, 9, 10, 11,
                                 12, 13, 14, 15);
}

// ---------------------------------------------------------------------------
__global__ void cvt_f32_f16(const float* __restrict__ in, h16* __restrict__ out,
                            int n) {
  int i = (blockIdx.x * 256 + threadIdx.x) * 4;
  if (i + 3 < n) {
    float4 f = *(const float4*)(in + i);
    out[i + 0] = (h16)f.x;
    out[i + 1] = (h16)f.y;
    out[i + 2] = (h16)f.z;
    out[i + 3] = (h16)f.w;
  }
}

// ---------------------------------------------------------------------------
// C[M,Nout] = A[M,K] @ W[Nout,K]^T + bias.  BM=BN=128, BK=32, 256 threads
// (8 wave32 waves in a 2x4 grid; each wave computes 64x32 via 8 WMMA accums).
// Double-buffered LDS: stage tile k+1 asynchronously while computing tile k.
template <bool HALF_OUT>
__global__ __launch_bounds__(256) void gemm_wmma(const h16* __restrict__ A,
                                                 const h16* __restrict__ W,
                                                 const float* __restrict__ bias,
                                                 void* __restrict__ Cv, int M,
                                                 int Nout, int K) {
  constexpr int BM = 128, BN = 128, BK = 32, LD = BK + 8;
  __shared__ __align__(16) h16 As[2][BM * LD];
  __shared__ __align__(16) h16 Bs[2][BN * LD];

  const int t = threadIdx.x;
  const int m0 = blockIdx.y * BM;
  const int n0 = blockIdx.x * BN;
  const int w = t >> 5;
  const int wm = (w >> 2) * 64;  // wave row offset in tile
  const int wn = (w & 3) * 32;   // wave col offset in tile

  // 128x32 f16 tile = 512 b128 chunks; 256 threads x 2 chunks each.
  auto stage = [&](h16* as, h16* bs, int k0) {
#pragma unroll
    for (int part = 0; part < 2; ++part) {
      int idx = t + part * 256;
      int row = idx >> 2;
      int c8 = (idx & 3) * 8;
      copy_b128_to_lds(&A[(size_t)(m0 + row) * K + k0 + c8], &as[row * LD + c8]);
      copy_b128_to_lds(&W[(size_t)(n0 + row) * K + k0 + c8], &bs[row * LD + c8]);
    }
  };

  v8f zero = {};
  v8f acc[4][2];
#pragma unroll
  for (int i = 0; i < 4; ++i)
#pragma unroll
    for (int j = 0; j < 2; ++j) acc[i][j] = zero;

  stage(As[0], Bs[0], 0);
  wait_async_copies();
  __syncthreads();

  int p = 0;
  for (int k0 = 0; k0 < K; k0 += BK) {
    // Kick off the next tile's async copies before computing this one.
    if (k0 + BK < K) stage(As[p ^ 1], Bs[p ^ 1], k0 + BK);

    v16h af[4], bf[2];
#pragma unroll
    for (int i = 0; i < 4; ++i)
      af[i] = load_frag_a(&As[p][(wm + i * 16) * LD], LD);
#pragma unroll
    for (int j = 0; j < 2; ++j)
      bf[j] = load_frag_b(&Bs[p][(wn + j * 16) * LD], LD);
#pragma unroll
    for (int i = 0; i < 4; ++i)
#pragma unroll
      for (int j = 0; j < 2; ++j)
        acc[i][j] = WMMA_F32_F16(af[i], bf[j], acc[i][j]);

    wait_async_copies();
    __syncthreads();
    p ^= 1;
  }

  // Epilogue: C/D layout — lane = col (n = lane&15), VGPR r = row (+8 high half)
  const int lane = t & 31;
  const int ncol = lane & 15;
  const int rbase = (lane & 16) ? 8 : 0;
#pragma unroll
  for (int j = 0; j < 2; ++j) {
    int n = n0 + wn + j * 16 + ncol;
    float bv = bias[n];
#pragma unroll
    for (int i = 0; i < 4; ++i) {
#pragma unroll
      for (int r = 0; r < 8; ++r) {
        size_t m = (size_t)(m0 + wm + i * 16 + rbase + r);
        float v = acc[i][j][r] + bv;
        if (HALF_OUT)
          ((h16*)Cv)[m * Nout + n] = (h16)v;
        else
          ((float*)Cv)[m * Nout + n] = v;
      }
    }
  }
}

// ---------------------------------------------------------------------------
// Flash-style attention. Block = one (b,h) x 128 query rows; 8 waves x 16 rows.
// qkv layout: [B*N, 3C] f16; q cols [0,768), k cols [768,1536), v cols [1536,2304).
__global__ __launch_bounds__(256) void attn_wmma(const h16* __restrict__ qkv,
                                                 h16* __restrict__ out) {
  constexpr int N = 1024, C = 768, C3 = 2304, HD = 96;
  constexpr float SCALE = 0.10206207261596577f;  // 96^-0.5

  __shared__ __align__(16) h16 Ks[32 * HD];     // [key][d]  row-major
  __shared__ __align__(16) h16 VTs[HD * 32];    // [d][key]  transposed
  __shared__ __align__(16) h16 Ps[8][16 * 32];  // per-wave P staging

  const int t = threadIdx.x;
  const int lane = t & 31;
  const int w = t >> 5;
  const int b = blockIdx.x >> 3;
  const int h = blockIdx.x & 7;
  const int q0 = blockIdx.y * 128 + w * 16;
  const size_t baseRow = (size_t)b * N;

  // Q fragments for this wave's 16 rows: hd=96 -> 3 chunks of K=32
  const h16* qbase = qkv + (baseRow + q0) * C3 + h * HD;
  v16h aq[3];
#pragma unroll
  for (int c = 0; c < 3; ++c) aq[c] = load_frag_a(qbase + c * 32, C3);

  v8f zero = {};
  v8f oacc[6];  // 16 x 96 output accumulator (6 n-tiles of 16)
#pragma unroll
  for (int g = 0; g < 6; ++g) oacc[g] = zero;
  float mrow[8], lrow[8];
#pragma unroll
  for (int r = 0; r < 8; ++r) {
    mrow[r] = -1e30f;
    lrow[r] = 0.0f;
  }

  const int ncol = lane & 15;
  const int rbase = (lane & 16) ? 8 : 0;

  for (int j0 = 0; j0 < N; j0 += 32) {
    __syncthreads();
    // K tile: async global->LDS (no layout change needed). 384 b128 chunks.
    for (int idx = t; idx < 32 * (HD / 8); idx += 256) {
      int row = idx / (HD / 8);
      int c8 = (idx % (HD / 8)) * 8;
      const h16* kp = qkv + (baseRow + j0 + row) * C3 + C + h * HD + c8;
      copy_b128_to_lds(kp, &Ks[row * HD + c8]);
      // V tile needs a transpose, so it goes through VGPRs.
      const h16* vp = qkv + (baseRow + j0 + row) * C3 + 2 * C + h * HD + c8;
      v8h vv = *(const v8h*)vp;
#pragma unroll
      for (int e = 0; e < 8; ++e) VTs[(c8 + e) * 32 + row] = vv[e];
    }
    wait_async_copies();
    __syncthreads();

    // S = Q @ K^T for 32 keys: load all 6 B-frags, then 6 back-to-back WMMAs.
    v16h bk[6];
#pragma unroll
    for (int c = 0; c < 3; ++c) {
      bk[c] = load_frag_b(&Ks[0 * HD + c * 32], HD);
      bk[3 + c] = load_frag_b(&Ks[16 * HD + c * 32], HD);
    }
    v8f s0 = zero, s1 = zero;
#pragma unroll
    for (int c = 0; c < 3; ++c) {
      s0 = WMMA_F32_F16(aq[c], bk[c], s0);
      s1 = WMMA_F32_F16(aq[c], bk[3 + c], s1);
    }

    // Online softmax (fp32). Rows 0-7 live in lanes 0-15, rows 8-15 in 16-31;
    // xor shuffles with mask<16 reduce within each half independently.
#pragma unroll
    for (int r = 0; r < 8; ++r) {
      float a0 = s0[r] * SCALE;
      float a1 = s1[r] * SCALE;
      float tm = fmaxf(a0, a1);
#pragma unroll
      for (int off = 1; off < 16; off <<= 1)
        tm = fmaxf(tm, __shfl_xor(tm, off, 32));
      float mnew = fmaxf(mrow[r], tm);
      float corr = __expf(mrow[r] - mnew);
      float p0 = __expf(a0 - mnew);
      float p1 = __expf(a1 - mnew);
      float rs = p0 + p1;
#pragma unroll
      for (int off = 1; off < 16; off <<= 1) rs += __shfl_xor(rs, off, 32);
      lrow[r] = lrow[r] * corr + rs;
      mrow[r] = mnew;
#pragma unroll
      for (int g = 0; g < 6; ++g) oacc[g][r] = oacc[g][r] * corr;
      // Stage P (f16) for re-striping into A-fragment layout.
      int m = rbase + r;
      Ps[w][m * 32 + ncol] = (h16)p0;
      Ps[w][m * 32 + 16 + ncol] = (h16)p1;
    }

    // O += P @ V: load all 6 V^T B-frags, then 6 back-to-back WMMAs.
    v16h ap = load_frag_a(&Ps[w][0], 32);
    v16h bv[6];
#pragma unroll
    for (int g = 0; g < 6; ++g) bv[g] = load_frag_b(&VTs[(g * 16) * 32], 32);
#pragma unroll
    for (int g = 0; g < 6; ++g) oacc[g] = WMMA_F32_F16(ap, bv[g], oacc[g]);
  }

  // Normalize and write [B*N, C] f16, head h occupies cols [h*96, h*96+96).
#pragma unroll
  for (int g = 0; g < 6; ++g) {
    int col = h * HD + g * 16 + ncol;
#pragma unroll
    for (int r = 0; r < 8; ++r) {
      size_t row = baseRow + q0 + rbase + r;
      float v = oacc[g][r] / lrow[r];
      out[row * C + col] = (h16)v;
    }
  }
}

// ---------------------------------------------------------------------------
extern "C" void kernel_launch(void* const* d_in, const int* in_sizes, int n_in,
                              void* d_out, int out_size, void* d_ws,
                              size_t ws_size, hipStream_t stream) {
  (void)in_sizes; (void)n_in; (void)out_size; (void)ws_size;
  const float* x      = (const float*)d_in[0];  // [16,1024,768]
  const float* w_qkv  = (const float*)d_in[1];  // [2304,768]
  const float* b_qkv  = (const float*)d_in[2];  // [2304]
  const float* w_proj = (const float*)d_in[3];  // [768,768]
  const float* b_proj = (const float*)d_in[4];  // [768]
  // d_in[5], d_in[6] = H, W (unused: bias path inactive)

  constexpr int B = 16, N = 1024, C = 768, C3 = 2304;
  const size_t M = (size_t)B * N;  // 16384

  char* ws = (char*)d_ws;
  h16* xh     = (h16*)ws; ws += M * C * sizeof(h16);           // 25.2 MB
  h16* wqkvh  = (h16*)ws; ws += (size_t)C3 * C * sizeof(h16);  //  3.5 MB
  h16* wprojh = (h16*)ws; ws += (size_t)C * C * sizeof(h16);   //  1.2 MB
  h16* qkvh   = (h16*)ws; ws += M * C3 * sizeof(h16);          // 75.5 MB
  h16* attnh  = (h16*)ws;                                      // 25.2 MB

  {
    int n = (int)(M * C);  // 12,582,912 -> exact grid
    cvt_f32_f16<<<dim3(n / 4 / 256), 256, 0, stream>>>(x, xh, n);
    n = C3 * C;
    cvt_f32_f16<<<dim3(n / 4 / 256), 256, 0, stream>>>(w_qkv, wqkvh, n);
    n = C * C;
    cvt_f32_f16<<<dim3(n / 4 / 256), 256, 0, stream>>>(w_proj, wprojh, n);
  }

  // QKV projection: [16384,768] @ [2304,768]^T -> f16 [16384,2304]
  gemm_wmma<true><<<dim3(C3 / 128, (int)(M / 128)), 256, 0, stream>>>(
      xh, wqkvh, b_qkv, qkvh, (int)M, C3, C);

  // Attention: grid = (B*nh, N/128)
  attn_wmma<<<dim3(B * 8, N / 128), 256, 0, stream>>>(qkvh, attnh);

  // Output projection: [16384,768] @ [768,768]^T + b_proj -> fp32 d_out
  gemm_wmma<false><<<dim3(C / 128, (int)(M / 128)), 256, 0, stream>>>(
      attnh, wprojh, b_proj, d_out, (int)M, C, C);
}
